// biGRU_attn_55198919688269
// MI455X (gfx1250) — compile-verified, matrix operand order checked by reading, not run on hardware
//
#include <hip/hip_runtime.h>
#include <hip/hip_bf16.h>

#define B_    16
#define SENC_ 512
#define TDEC_ 64
#define E_    512
#define H_    512
#define V_    32000

typedef __attribute__((ext_vector_type(16))) __bf16 v16bf;
typedef __attribute__((ext_vector_type(8)))  float  v8f;

union Frag16 { v16bf v; unsigned short u[16]; };

__device__ __forceinline__ unsigned short f32_to_bf16(float f) {
  unsigned int u = __float_as_uint(f);
  unsigned int r = (u + 0x7FFFu + ((u >> 16) & 1u)) >> 16;
  return (unsigned short)r;
}

// A/B fragment for v_wmma_f32_16x16x32_bf16.
// 16-bit A 16x32 layout (ISA 7.12.2): lane l: row = l&15, half h = l>>4.
// VGPR0..3 hold K = 8h+0..7 (contiguous), VGPR4..7 hold K = 16+8h+0..7 (contiguous)
// -> two 16-byte contiguous loads per lane. B loads identically from W[N][K]
// (torch layout), which is exactly B^T, giving C = A * W^T.
__device__ __forceinline__ v16bf load_frag_bf16(const unsigned short* base, int ld, int k0) {
  const int lane = threadIdx.x & 31;
  const int row  = lane & 15;
  const int h8   = (lane >> 4) << 3;
  const unsigned short* p = base + (size_t)row * ld + k0 + h8;
  Frag16 f;
#pragma unroll
  for (int i = 0; i < 8; ++i) { f.u[i] = p[i]; f.u[8 + i] = p[16 + i]; }
  return f.v;
}

__device__ __forceinline__ v16bf load_frag_f32(const float* base, size_t ld, int k0) {
  const int lane = threadIdx.x & 31;
  const int row  = lane & 15;
  const int h8   = (lane >> 4) << 3;
  const float* p = base + (size_t)row * ld + k0 + h8;
  Frag16 f;
#pragma unroll
  for (int i = 0; i < 8; ++i) { f.u[i] = f32_to_bf16(p[i]); f.u[8 + i] = f32_to_bf16(p[16 + i]); }
  return f.v;
}

#define WMMA_BF16(a, b, c) \
  __builtin_amdgcn_wmma_f32_16x16x32_bf16(false, (a), false, (b), (short)0, (c), false, false)

// ---------------------------------------------------------------------------
// fp32 -> bf16 conversion
__global__ __launch_bounds__(256) void cvt_bf16(const float* __restrict__ s,
                                                unsigned short* __restrict__ d, int n) {
  int i = blockIdx.x * 256 + threadIdx.x;
  if (i < n) d[i] = f32_to_bf16(s[i]);
}

// embedding gather -> bf16 rows (row = flat token index)
__global__ __launch_bounds__(256) void gather_enc(const int* __restrict__ tok,
                                                  const float* __restrict__ embed,
                                                  unsigned short* __restrict__ out, int nrow) {
  int i = blockIdx.x * 256 + threadIdx.x;
  if (i >= nrow * E_) return;
  int r = i >> 9, e = i & (E_ - 1);
  out[i] = f32_to_bf16(embed[(size_t)tok[r] * E_ + e]);
}

// decoder embedding gather with SOS shift: token(t=0)=SOS(=1), else dec_in[t-1]
__global__ __launch_bounds__(256) void gather_dec(const int* __restrict__ dec_in,
                                                  const float* __restrict__ embed,
                                                  unsigned short* __restrict__ out) {
  int i = blockIdx.x * 256 + threadIdx.x;
  if (i >= B_ * TDEC_ * E_) return;
  int r = i >> 9, e = i & (E_ - 1);
  int b = r >> 6, t = r & (TDEC_ - 1);
  int tok = (t == 0) ? 1 : dec_in[b * TDEC_ + t - 1];
  out[i] = f32_to_bf16(embed[(size_t)tok * E_ + e]);
}

// ---------------------------------------------------------------------------
// C[M,N] = A[M,K](bf16) * W[N,K](bf16)^T + bias[N]; 8 waves, wave tile 16x64.
// Software-pipelined: fragments for k-step (k0+32) are loaded while the four
// WMMAs for k0 execute, so loads overlap matrix math instead of s_wait 0.
__global__ __launch_bounds__(256) void gemm_bf16_bias(
    const unsigned short* __restrict__ A, const unsigned short* __restrict__ W,
    const float* __restrict__ bias, float* __restrict__ C, int M, int N, int K) {
  const int wave = threadIdx.x >> 5;
  const int m0 = blockIdx.y * 32 + ((wave >> 2) << 4);
  const int n0 = blockIdx.x * 256 + ((wave & 3) << 6);
  if (m0 >= M || n0 >= N) return;
  const unsigned short* Abase = A + (size_t)m0 * K;
  const unsigned short* W0 = W + (size_t)(n0 + 0)  * K;
  const unsigned short* W1 = W + (size_t)(n0 + 16) * K;
  const unsigned short* W2 = W + (size_t)(n0 + 32) * K;
  const unsigned short* W3 = W + (size_t)(n0 + 48) * K;

  v8f acc[4] = {};
  v16bf a_c  = load_frag_bf16(Abase, K, 0);
  v16bf b0_c = load_frag_bf16(W0, K, 0);
  v16bf b1_c = load_frag_bf16(W1, K, 0);
  v16bf b2_c = load_frag_bf16(W2, K, 0);
  v16bf b3_c = load_frag_bf16(W3, K, 0);
  for (int k0 = 0; k0 < K; k0 += 32) {
    const int kn = (k0 + 32 < K) ? (k0 + 32) : 0;  // wrap: harmless dead prefetch
    v16bf a_n  = load_frag_bf16(Abase, K, kn);
    v16bf b0_n = load_frag_bf16(W0, K, kn);
    v16bf b1_n = load_frag_bf16(W1, K, kn);
    v16bf b2_n = load_frag_bf16(W2, K, kn);
    v16bf b3_n = load_frag_bf16(W3, K, kn);
    acc[0] = WMMA_BF16(a_c, b0_c, acc[0]);
    acc[1] = WMMA_BF16(a_c, b1_c, acc[1]);
    acc[2] = WMMA_BF16(a_c, b2_c, acc[2]);
    acc[3] = WMMA_BF16(a_c, b3_c, acc[3]);
    a_c = a_n; b0_c = b0_n; b1_c = b1_n; b2_c = b2_n; b3_c = b3_n;
  }
  const int lane = threadIdx.x & 31;
  const int nn = lane & 15;
  const int mv = (lane >> 4) << 3;
#pragma unroll
  for (int t = 0; t < 4; ++t) {
    int nc = n0 + t * 16 + nn;
    float bv = bias ? bias[nc] : 0.0f;
#pragma unroll
    for (int v = 0; v < 8; ++v)
      C[(size_t)(m0 + mv + v) * N + nc] = acc[t][v] + bv;
  }
}

// ---------------------------------------------------------------------------
// Persistent per-direction GRU scan: grid.x = 2 (fwd/bwd), 1024 threads = 32 waves.
// Wave j owns hidden columns [16j, 16j+16); computes r,z,n gate tiles for those
// columns via WMMA (K=512, A = h in LDS), so the gate mix stays in registers.
__global__ __launch_bounds__(1024) void enc_recurrent(
    const float* __restrict__ Xf, const float* __restrict__ Xb,
    const unsigned short* __restrict__ Whhf, const unsigned short* __restrict__ Whhb,
    const float* __restrict__ bhhf, const float* __restrict__ bhhb,
    float* __restrict__ encoded) {
  __shared__ unsigned short h_bf[B_ * H_];  // 16KB
  __shared__ float          h_f [B_ * H_];  // 32KB
  const int dir = blockIdx.x;
  const float* X = dir ? Xb : Xf;
  const unsigned short* Whh = dir ? Whhb : Whhf;
  const float* bhh = dir ? bhhb : bhhf;
  for (int i = threadIdx.x; i < B_ * H_; i += 1024) { h_bf[i] = 0; h_f[i] = 0.f; }
  __syncthreads();
  const int wave = threadIdx.x >> 5;
  const int lane = threadIdx.x & 31;
  const int nn = lane & 15, m0 = (lane >> 4) << 3;
  const int jc = (wave << 4) + nn;
  const float bhr = bhh[jc], bhz = bhh[H_ + jc], bhn = bhh[2 * H_ + jc];
  const unsigned short* Wr = Whh + (size_t)(0 * H_ + (wave << 4)) * H_;
  const unsigned short* Wz = Whh + (size_t)(1 * H_ + (wave << 4)) * H_;
  const unsigned short* Wn = Whh + (size_t)(2 * H_ + (wave << 4)) * H_;
  for (int step = 0; step < SENC_; ++step) {
    const int s = dir ? (SENC_ - 1 - step) : step;
    v8f ar = {}, az = {}, an = {};
    // pipelined K loop: prefetch (k0+32) fragments while k0 WMMAs run
    v16bf a_c  = load_frag_bf16(h_bf, H_, 0);
    v16bf br_c = load_frag_bf16(Wr, H_, 0);
    v16bf bz_c = load_frag_bf16(Wz, H_, 0);
    v16bf bn_c = load_frag_bf16(Wn, H_, 0);
    for (int k0 = 0; k0 < H_; k0 += 32) {
      const int kn = (k0 + 32) & (H_ - 1);
      v16bf a_n  = load_frag_bf16(h_bf, H_, kn);
      v16bf br_n = load_frag_bf16(Wr, H_, kn);
      v16bf bz_n = load_frag_bf16(Wz, H_, kn);
      v16bf bn_n = load_frag_bf16(Wn, H_, kn);
      ar = WMMA_BF16(a_c, br_c, ar);
      az = WMMA_BF16(a_c, bz_c, az);
      an = WMMA_BF16(a_c, bn_c, an);
      a_c = a_n; br_c = br_n; bz_c = bz_n; bn_c = bn_n;
    }
    __syncthreads();  // all reads of h done before updates
#pragma unroll
    for (int v = 0; v < 8; ++v) {
      const int b = m0 + v;
      const size_t xrow = ((size_t)b * SENC_ + s) * (3 * H_);
      const float gir = X[xrow + jc];
      const float giz = X[xrow + H_ + jc];
      const float gin = X[xrow + 2 * H_ + jc];
      const float r = 1.f / (1.f + __expf(-(gir + ar[v] + bhr)));
      const float z = 1.f / (1.f + __expf(-(giz + az[v] + bhz)));
      const float n = tanhf(gin + r * (an[v] + bhn));
      const float hnew = (1.f - z) * n + z * h_f[b * H_ + jc];
      h_f[b * H_ + jc] = hnew;
      h_bf[b * H_ + jc] = f32_to_bf16(hnew);
      encoded[((size_t)b * SENC_ + s) * (2 * H_) + dir * H_ + jc] = hnew;
    }
    __syncthreads();
  }
}

// ---------------------------------------------------------------------------
// Persistent decoder: 1 block, 1024 threads = 32 waves, 64 steps.
__global__ __launch_bounds__(1024) void decoder_attn(
    const float* __restrict__ encoded,
    const unsigned short* __restrict__ dec_emb,  // [B,T,E] bf16
    const unsigned short* __restrict__ Wih,      // [1536,1536] bf16
    const unsigned short* __restrict__ Whh,      // [1536,512]  bf16
    const float* __restrict__ bih, const float* __restrict__ bhh,
    const unsigned short* __restrict__ W1bf, const float* __restrict__ W1b,
    const unsigned short* __restrict__ W2bf, const float* __restrict__ W2b,
    float* __restrict__ comp,   // [16,1024] global scratch
    float* __restrict__ attn,   // [16,512]  global scratch
    unsigned short* __restrict__ outs) {  // [B*T, H] bf16
  __shared__ unsigned short ctx_bf[B_ * 2 * H_];  // 32KB
  __shared__ unsigned short h_bf  [B_ * H_];      // 16KB
  __shared__ float red[32];
  __shared__ float rowinv[16];
  const int tid = threadIdx.x;
  const int wave = tid >> 5;
  const int lane = tid & 31;
  const int nn = lane & 15, m0 = (lane >> 4) << 3;
  const int jc = (wave << 4) + nn;

  for (int i = tid; i < B_ * 2 * H_; i += 1024) ctx_bf[i] = 0;  // ctx0 = 0

  // h0 = encoded[:, -1, :] @ W1^T + W1_b  (K = 1024)
  float hreg[8];
  {
    v8f acc = {};
    const float* Alast = encoded + (size_t)(SENC_ - 1) * (2 * H_);
    const unsigned short* Wp = W1bf + (size_t)(wave << 4) * (2 * H_);
    for (int k0 = 0; k0 < 2 * H_; k0 += 32) {
      v16bf a = load_frag_f32(Alast, (size_t)SENC_ * (2 * H_), k0);
      acc = WMMA_BF16(a, load_frag_bf16(Wp, 2 * H_, k0), acc);
    }
    const float bb = W1b[jc];
#pragma unroll
    for (int v = 0; v < 8; ++v) {
      hreg[v] = acc[v] + bb;
      h_bf[(m0 + v) * H_ + jc] = f32_to_bf16(hreg[v]);
    }
  }
  __syncthreads();

  const float b_r  = bih[jc] + bhh[jc];
  const float b_z  = bih[H_ + jc] + bhh[H_ + jc];
  const float b_in = bih[2 * H_ + jc];
  const float b_hn = bhh[2 * H_ + jc];
  const int KIN = 2 * H_ + E_;  // 1536

  const unsigned short* Wr = Whh + (size_t)(0 * H_ + (wave << 4)) * H_;
  const unsigned short* Wz = Whh + (size_t)(1 * H_ + (wave << 4)) * H_;
  const unsigned short* Wn = Whh + (size_t)(2 * H_ + (wave << 4)) * H_;
  const unsigned short* Ur = Wih + (size_t)(0 * H_ + (wave << 4)) * KIN;
  const unsigned short* Uz = Wih + (size_t)(1 * H_ + (wave << 4)) * KIN;
  const unsigned short* Un = Wih + (size_t)(2 * H_ + (wave << 4)) * KIN;

  for (int t = 0; t < TDEC_; ++t) {
    // -------- stage A: GRU cell (gi over ctx||emb, gh over h) --------
    v8f ar = {}, az = {}, ain = {}, ahn = {};
    {  // gh: K = 512, A = h (LDS); pipelined
      v16bf a_c  = load_frag_bf16(h_bf, H_, 0);
      v16bf br_c = load_frag_bf16(Wr, H_, 0);
      v16bf bz_c = load_frag_bf16(Wz, H_, 0);
      v16bf bn_c = load_frag_bf16(Wn, H_, 0);
      for (int k0 = 0; k0 < H_; k0 += 32) {
        const int kn = (k0 + 32) & (H_ - 1);
        v16bf a_n  = load_frag_bf16(h_bf, H_, kn);
        v16bf br_n = load_frag_bf16(Wr, H_, kn);
        v16bf bz_n = load_frag_bf16(Wz, H_, kn);
        v16bf bn_n = load_frag_bf16(Wn, H_, kn);
        ar  = WMMA_BF16(a_c, br_c, ar);
        az  = WMMA_BF16(a_c, bz_c, az);
        ahn = WMMA_BF16(a_c, bn_c, ahn);
        a_c = a_n; br_c = br_n; bz_c = bz_n; bn_c = bn_n;
      }
    }
    {  // gi part 1: K in [0,1024), A = ctx (LDS); pipelined
      v16bf a_c  = load_frag_bf16(ctx_bf, 2 * H_, 0);
      v16bf br_c = load_frag_bf16(Ur, KIN, 0);
      v16bf bz_c = load_frag_bf16(Uz, KIN, 0);
      v16bf bn_c = load_frag_bf16(Un, KIN, 0);
      for (int k0 = 0; k0 < 2 * H_; k0 += 32) {
        const int kn = (k0 + 32) & (2 * H_ - 1);
        v16bf a_n  = load_frag_bf16(ctx_bf, 2 * H_, kn);
        v16bf br_n = load_frag_bf16(Ur, KIN, kn);
        v16bf bz_n = load_frag_bf16(Uz, KIN, kn);
        v16bf bn_n = load_frag_bf16(Un, KIN, kn);
        ar  = WMMA_BF16(a_c, br_c, ar);
        az  = WMMA_BF16(a_c, bz_c, az);
        ain = WMMA_BF16(a_c, bn_c, ain);
        a_c = a_n; br_c = br_n; bz_c = bz_n; bn_c = bn_n;
      }
    }
    {  // gi part 2: K in [1024,1536), A = dec_emb (global); pipelined
      const unsigned short* Eb = dec_emb + (size_t)t * E_;  // row stride T*E
      v16bf a_c  = load_frag_bf16(Eb, TDEC_ * E_, 0);
      v16bf br_c = load_frag_bf16(Ur, KIN, 2 * H_);
      v16bf bz_c = load_frag_bf16(Uz, KIN, 2 * H_);
      v16bf bn_c = load_frag_bf16(Un, KIN, 2 * H_);
      for (int k0 = 0; k0 < E_; k0 += 32) {
        const int kn = (k0 + 32) & (E_ - 1);
        v16bf a_n  = load_frag_bf16(Eb, TDEC_ * E_, kn);
        v16bf br_n = load_frag_bf16(Ur, KIN, 2 * H_ + kn);
        v16bf bz_n = load_frag_bf16(Uz, KIN, 2 * H_ + kn);
        v16bf bn_n = load_frag_bf16(Un, KIN, 2 * H_ + kn);
        ar  = WMMA_BF16(a_c, br_c, ar);
        az  = WMMA_BF16(a_c, bz_c, az);
        ain = WMMA_BF16(a_c, bn_c, ain);
        a_c = a_n; br_c = br_n; bz_c = bz_n; bn_c = bn_n;
      }
    }
    __syncthreads();  // all h_bf/ctx_bf reads complete
#pragma unroll
    for (int v = 0; v < 8; ++v) {
      const int b = m0 + v;
      const float r = 1.f / (1.f + __expf(-(ar[v] + b_r)));
      const float z = 1.f / (1.f + __expf(-(az[v] + b_z)));
      const float n = tanhf(ain[v] + b_in + r * (ahn[v] + b_hn));
      const float hnew = (1.f - z) * n + z * hreg[v];
      hreg[v] = hnew;
      h_bf[b * H_ + jc] = f32_to_bf16(hnew);
      outs[(size_t)(b * TDEC_ + t) * H_ + jc] = f32_to_bf16(hnew);
    }
    __syncthreads();

    // -------- stage B: comp = h_new @ W2^T + W2_b  (2 tiles / wave) --------
    {
      const unsigned short* P0 = W2bf + (size_t)((wave << 5) + 0)  * H_;
      const unsigned short* P1 = W2bf + (size_t)((wave << 5) + 16) * H_;
      v8f c2[2] = {};
      v16bf a_c  = load_frag_bf16(h_bf, H_, 0);
      v16bf b0_c = load_frag_bf16(P0, H_, 0);
      v16bf b1_c = load_frag_bf16(P1, H_, 0);
      for (int k0 = 0; k0 < H_; k0 += 32) {
        const int kn = (k0 + 32) & (H_ - 1);
        v16bf a_n  = load_frag_bf16(h_bf, H_, kn);
        v16bf b0_n = load_frag_bf16(P0, H_, kn);
        v16bf b1_n = load_frag_bf16(P1, H_, kn);
        c2[0] = WMMA_BF16(a_c, b0_c, c2[0]);
        c2[1] = WMMA_BF16(a_c, b1_c, c2[1]);
        a_c = a_n; b0_c = b0_n; b1_c = b1_n;
      }
#pragma unroll
      for (int i = 0; i < 2; ++i) {
        const int nc = (wave << 5) + (i << 4) + nn;
        const float bb = W2b[nc];
#pragma unroll
        for (int v = 0; v < 8; ++v)
          comp[(size_t)(m0 + v) * (2 * H_) + nc] = c2[i][v] + bb;
      }
    }
    __threadfence();
    __syncthreads();

    // -------- stage C: scores = encoded . comp, softmax over s --------
#pragma unroll
    for (int i = 0; i < 8; ++i) {
      const int p = tid + (i << 10);
      const int b = p >> 9, s = p & (SENC_ - 1);
      const float* er = encoded + ((size_t)b * SENC_ + s) * (2 * H_);
      const float* cb = comp + (size_t)b * (2 * H_);
      float acc = 0.f;
      for (int k = 0; k < 2 * H_; k += 4) {
        const float4 ev = *(const float4*)(er + k);
        const float4 cv = *(const float4*)(cb + k);
        acc += ev.x * cv.x + ev.y * cv.y + ev.z * cv.z + ev.w * cv.w;
      }
      attn[p] = acc;
    }
    __threadfence();
    __syncthreads();
    {  // softmax: 2 waves per batch row
      const int row = wave >> 1;
      const int base = row * SENC_ + ((wave & 1) << 8) + lane;
      float mx = -3.4e38f;
#pragma unroll
      for (int i = 0; i < 8; ++i) mx = fmaxf(mx, attn[base + (i << 5)]);
#pragma unroll
      for (int o = 16; o > 0; o >>= 1) mx = fmaxf(mx, __shfl_xor(mx, o));
      if (lane == 0) red[wave] = mx;
      __syncthreads();
      const float rmax = fmaxf(red[row << 1], red[(row << 1) + 1]);
      __syncthreads();
      float sum = 0.f;
#pragma unroll
      for (int i = 0; i < 8; ++i) {
        const int idx = base + (i << 5);
        const float e = __expf(attn[idx] - rmax);
        attn[idx] = e;
        sum += e;
      }
#pragma unroll
      for (int o = 16; o > 0; o >>= 1) sum += __shfl_xor(sum, o);
      if (lane == 0) red[wave] = sum;
      __threadfence();
      __syncthreads();
      if (lane == 0 && (wave & 1) == 0)
        rowinv[row] = 1.f / (red[row << 1] + red[(row << 1) + 1]);
    }
    __syncthreads();

    // -------- stage D: ctx = (attn @ encoded) * rowinv -> bf16 LDS --------
    for (int b = 0; b < B_; ++b) {
      const float* er = encoded + ((size_t)b * SENC_) * (2 * H_) + tid;
      const float* ap = attn + b * SENC_;
      float acc = 0.f;
      for (int s = 0; s < SENC_; ++s) acc += ap[s] * er[(size_t)s * (2 * H_)];
      ctx_bf[b * (2 * H_) + tid] = f32_to_bf16(acc * rowinv[b]);
    }
    __syncthreads();
  }
}

// ---------------------------------------------------------------------------
extern "C" void kernel_launch(void* const* d_in, const int* in_sizes, int n_in,
                              void* d_out, int out_size, void* d_ws, size_t ws_size,
                              hipStream_t stream) {
  (void)in_sizes; (void)n_in; (void)out_size; (void)ws_size;
  const int*   enc_in = (const int*)d_in[0];
  const int*   dec_in = (const int*)d_in[1];
  const float* embed  = (const float*)d_in[2];
  const float* Wih_f  = (const float*)d_in[3];
  const float* Whh_f  = (const float*)d_in[4];
  const float* bih_f  = (const float*)d_in[5];
  const float* bhh_f  = (const float*)d_in[6];
  const float* Wih_b  = (const float*)d_in[7];
  const float* Whh_b  = (const float*)d_in[8];
  const float* bih_b  = (const float*)d_in[9];
  const float* bhh_b  = (const float*)d_in[10];
  const float* dWih   = (const float*)d_in[11];
  const float* dWhh   = (const float*)d_in[12];
  const float* dbih   = (const float*)d_in[13];
  const float* dbhh   = (const float*)d_in[14];
  const float* W1w    = (const float*)d_in[15];
  const float* W1b    = (const float*)d_in[16];
  const float* W2w    = (const float*)d_in[17];
  const float* W2b    = (const float*)d_in[18];
  const float* linw   = (const float*)d_in[19];
  const float* linb   = (const float*)d_in[20];
  float* out = (float*)d_out;

  char* ws = (char*)d_ws;
  size_t off = 0;
  auto take = [&](size_t bytes) -> void* {
    void* p = ws + off;
    off += (bytes + 255) & ~(size_t)255;
    return p;
  };
  unsigned short* enc_emb = (unsigned short*)take((size_t)B_ * SENC_ * E_ * 2);
  unsigned short* dec_emb = (unsigned short*)take((size_t)B_ * TDEC_ * E_ * 2);
  unsigned short* wihf_bf = (unsigned short*)take((size_t)3 * H_ * E_ * 2);
  unsigned short* wihb_bf = (unsigned short*)take((size_t)3 * H_ * E_ * 2);
  unsigned short* whhf_bf = (unsigned short*)take((size_t)3 * H_ * H_ * 2);
  unsigned short* whhb_bf = (unsigned short*)take((size_t)3 * H_ * H_ * 2);
  unsigned short* dwih_bf = (unsigned short*)take((size_t)3 * H_ * (E_ + 2 * H_) * 2);
  unsigned short* dwhh_bf = (unsigned short*)take((size_t)3 * H_ * H_ * 2);
  unsigned short* w1_bf   = (unsigned short*)take((size_t)H_ * 2 * H_ * 2);
  unsigned short* w2_bf   = (unsigned short*)take((size_t)2 * H_ * H_ * 2);
  unsigned short* lin_bf  = (unsigned short*)take((size_t)V_ * H_ * 2);
  unsigned short* outs_bf = (unsigned short*)take((size_t)B_ * TDEC_ * H_ * 2);
  float* Xf      = (float*)take((size_t)B_ * SENC_ * 3 * H_ * 4);
  float* Xb      = (float*)take((size_t)B_ * SENC_ * 3 * H_ * 4);
  float* encoded = (float*)take((size_t)B_ * SENC_ * 2 * H_ * 4);
  float* comp    = (float*)take((size_t)B_ * 2 * H_ * 4);
  float* attn    = (float*)take((size_t)B_ * SENC_ * 4);

  // 1) weights -> bf16
  struct { const float* s; unsigned short* d; int n; } jobs[9] = {
    {Wih_f, wihf_bf, 3 * H_ * E_}, {Wih_b, wihb_bf, 3 * H_ * E_},
    {Whh_f, whhf_bf, 3 * H_ * H_}, {Whh_b, whhb_bf, 3 * H_ * H_},
    {dWih, dwih_bf, 3 * H_ * (E_ + 2 * H_)}, {dWhh, dwhh_bf, 3 * H_ * H_},
    {W1w, w1_bf, H_ * 2 * H_}, {W2w, w2_bf, 2 * H_ * H_},
    {linw, lin_bf, V_ * H_},
  };
  for (int j = 0; j < 9; ++j)
    cvt_bf16<<<(jobs[j].n + 255) / 256, 256, 0, stream>>>(jobs[j].s, jobs[j].d, jobs[j].n);

  // 2) embedding gathers
  gather_enc<<<(B_ * SENC_ * E_ + 255) / 256, 256, 0, stream>>>(enc_in, embed, enc_emb, B_ * SENC_);
  gather_dec<<<(B_ * TDEC_ * E_ + 255) / 256, 256, 0, stream>>>(dec_in, embed, dec_emb);

  // 3) pre-gate GEMMs: X = enc_emb @ Wih^T + bih   [8192 x 1536, K=512]
  {
    dim3 g((3 * H_) / 256, (B_ * SENC_) / 32);
    gemm_bf16_bias<<<g, 256, 0, stream>>>(enc_emb, wihf_bf, bih_f, Xf, B_ * SENC_, 3 * H_, E_);
    gemm_bf16_bias<<<g, 256, 0, stream>>>(enc_emb, wihb_bf, bih_b, Xb, B_ * SENC_, 3 * H_, E_);
  }

  // 4) bidirectional GRU scan (both directions concurrently)
  enc_recurrent<<<2, 1024, 0, stream>>>(Xf, Xb, whhf_bf, whhb_bf, bhh_f, bhh_b, encoded);

  // 5) decoder with attention (64 sequential steps)
  decoder_attn<<<1, 1024, 0, stream>>>(encoded, dec_emb, dwih_bf, dwhh_bf, dbih, dbhh,
                                       w1_bf, W1b, w2_bf, W2b, comp, attn, outs_bf);

  // 6) logits = outs @ lin_w^T + lin_b   [1024 x 32000, K=512] -> d_out
  {
    dim3 g(V_ / 256, (B_ * TDEC_) / 32);
    gemm_bf16_bias<<<g, 256, 0, stream>>>(outs_bf, lin_bf, linb, out, B_ * TDEC_, V_, H_);
  }
}
